// PointNet2Seg_3650722202244
// MI455X (gfx1250) — matrix operand order repaired and checked by reading, not compiled, not run on hardware
//
#include <hip/hip_runtime.h>
#include <hip/hip_bf16.h>

// ---------------------------------------------------------------------------
// PointNet++ segmentation forward for MI455X (gfx1250, wave32).
// MLPs run as f16 WMMA GEMMs (v_wmma_f32_16x16x32_f16) with BN folded into a
// fused scale/bias epilogue. Weight tiles are staged block-wide into LDS via
// GLOBAL_LOAD_ASYNC_TO_LDS_B128 (ASYNCcnt-tracked DMA; ds fallback), A
// fragments stream from global with prefetch. Geometry is scalar f32.
// Workspace requirement ~80 MB.
// ---------------------------------------------------------------------------

typedef __attribute__((ext_vector_type(16))) _Float16 v16h;
typedef __attribute__((ext_vector_type(8)))  _Float16 v8h;
typedef __attribute__((ext_vector_type(8)))  float    v8f;
typedef __attribute__((ext_vector_type(4)))  int      v4i;

#define BN_EPS 1e-5f

#if defined(__HIP_DEVICE_COMPILE__) && \
    __has_builtin(__builtin_amdgcn_global_load_async_to_lds_b128) && \
    __has_builtin(__builtin_amdgcn_s_wait_asynccnt)
#define HAVE_ASYNC_LDS 1
#else
#define HAVE_ASYNC_LDS 0
#endif

// ---------------------------------------------------------------------------
// Weight prep: W(cout,cin) f32 -> Wh(NcPad,Kp) f16 zero-padded, plus fused
// BN scale/bias per output channel. g==nullptr => plain linear (scale=1,bias=b)
// ---------------------------------------------------------------------------
__global__ void prep_layer_kernel(const float* __restrict__ W, const float* __restrict__ b,
                                  const float* __restrict__ g, const float* __restrict__ bt,
                                  const float* __restrict__ m, const float* __restrict__ v,
                                  int cout, int cin, int Kp, int NcPad,
                                  _Float16* __restrict__ Wh,
                                  float* __restrict__ scale, float* __restrict__ bias) {
  long long total = (long long)NcPad * Kp;
  long long stride = (long long)gridDim.x * blockDim.x;
  for (long long e = (long long)blockIdx.x * blockDim.x + threadIdx.x; e < total; e += stride) {
    int n = (int)(e / Kp), k = (int)(e % Kp);
    float w = (n < cout && k < cin) ? W[(long long)n * cin + k] : 0.f;
    Wh[e] = (_Float16)w;
  }
  int gid = blockIdx.x * blockDim.x + threadIdx.x;
  if (gid < NcPad) {
    if (gid < cout) {
      if (g) {
        float s = g[gid] * rsqrtf(v[gid] + BN_EPS);
        scale[gid] = s;
        bias[gid]  = (b[gid] - m[gid]) * s + bt[gid];
      } else {
        scale[gid] = 1.f;
        bias[gid]  = b[gid];
      }
    } else {
      scale[gid] = 0.f;
      bias[gid]  = 0.f;
    }
  }
}

// ---------------------------------------------------------------------------
// Farthest point sampling: one block per batch. Matches jax.lax.scan semantics
// (collect farthest entering each step; first index is 0). Also gathers new_xyz.
// ---------------------------------------------------------------------------
__global__ void fps_kernel(const float* __restrict__ xyz, int xs, int N, int npoint,
                           int* __restrict__ out_idx, float* __restrict__ newxyz,
                           float* __restrict__ distws) {
  int b = blockIdx.x;
  const float* X = xyz + (long long)b * N * xs;
  float* dist = distws + (long long)b * N;
  int*   idx  = out_idx + (long long)b * npoint;
  float* nx   = newxyz + (long long)b * npoint * 3;
  int t = threadIdx.x, nt = blockDim.x;

  for (int i = t; i < N; i += nt) dist[i] = 1e10f;
  __syncthreads();

  __shared__ float s_best[256];
  __shared__ int   s_bidx[256];
  int farthest = 0;
  for (int it = 0; it < npoint; ++it) {
    if (t == 0) {
      idx[it] = farthest;
      const float* c = X + (long long)farthest * xs;
      nx[it * 3 + 0] = c[0]; nx[it * 3 + 1] = c[1]; nx[it * 3 + 2] = c[2];
    }
    float cx = X[(long long)farthest * xs + 0];
    float cy = X[(long long)farthest * xs + 1];
    float cz = X[(long long)farthest * xs + 2];
    float best = -1.f; int bidx = 0;
    for (int i = t; i < N; i += nt) {
      float dx = X[(long long)i * xs + 0] - cx;
      float dy = X[(long long)i * xs + 1] - cy;
      float dz = X[(long long)i * xs + 2] - cz;
      float d  = dx * dx + dy * dy + dz * dz;
      float dd = fminf(dist[i], d);
      dist[i] = dd;
      if (dd > best) { best = dd; bidx = i; }   // strictly greater => first-max wins
    }
    s_best[t] = best; s_bidx[t] = bidx;
    __syncthreads();
    for (int off = nt >> 1; off > 0; off >>= 1) {
      if (t < off) {
        bool take = (s_best[t + off] > s_best[t]) ||
                    (s_best[t + off] == s_best[t] && s_bidx[t + off] < s_bidx[t]);
        if (take) { s_best[t] = s_best[t + off]; s_bidx[t] = s_bidx[t + off]; }
      }
      __syncthreads();
    }
    farthest = s_bidx[0];
    __syncthreads();
  }
}

// ---------------------------------------------------------------------------
// Ball query: one thread per center; index-ordered scan == sort+take-first.
// ---------------------------------------------------------------------------
__global__ void ball_query_kernel(const float* __restrict__ xyz, int xs, int N,
                                  const float* __restrict__ newxyz, int npoint,
                                  float r2, int nsample, int* __restrict__ out, int B) {
  int gid = blockIdx.x * blockDim.x + threadIdx.x;
  if (gid >= B * npoint) return;
  int b = gid / npoint;
  const float* X = xyz + (long long)b * N * xs;
  const float* c = newxyz + (long long)gid * 3;
  int* o = out + (long long)gid * nsample;
  int cnt = 0, first = 0; bool have = false;
  for (int i = 0; i < N && cnt < nsample; ++i) {
    float dx = X[(long long)i * xs + 0] - c[0];
    float dy = X[(long long)i * xs + 1] - c[1];
    float dz = X[(long long)i * xs + 2] - c[2];
    if (dx * dx + dy * dy + dz * dz <= r2) {
      if (!have) { first = i; have = true; }
      o[cnt++] = i;
    }
  }
  for (int k = cnt; k < nsample; ++k) o[k] = first;
}

// ---------------------------------------------------------------------------
// Build grouped GEMM input: [points[idx] (Cp ch), xyz[idx]-center (3 ch), 0-pad]
// ---------------------------------------------------------------------------
__global__ void group_kernel(const float* __restrict__ xyz, int xs,
                             const _Float16* __restrict__ pts16,
                             const float* __restrict__ pts32, int Cp, int pstride,
                             const float* __restrict__ newxyz, const int* __restrict__ idx,
                             int npoint, int nsample, int N,
                             _Float16* __restrict__ out, int Kp, int B) {
  long long total = (long long)B * npoint * nsample * Kp;
  long long gs = (long long)gridDim.x * blockDim.x;
  for (long long e = (long long)blockIdx.x * blockDim.x + threadIdx.x; e < total; e += gs) {
    int k = (int)(e % Kp);
    long long r = e / Kp;
    int j = (int)(r % nsample);
    long long bs = r / nsample;
    int s = (int)(bs % npoint);
    int b = (int)(bs / npoint);
    int pi = idx[((long long)b * npoint + s) * nsample + j];
    float val = 0.f;
    if (k < Cp) {
      long long po = ((long long)b * N + pi) * pstride + k;
      val = pts16 ? (float)pts16[po] : pts32[po];
    } else if (k < Cp + 3) {
      int c = k - Cp;
      val = xyz[((long long)b * N + pi) * xs + c] - newxyz[((long long)b * npoint + s) * 3 + c];
    }
    out[e] = (_Float16)val;
  }
}

// ---------------------------------------------------------------------------
// WMMA GEMM + fused BN/ReLU epilogue. X:(M,Kp) f16 row-major, Wh:(NcPad,Kp)
// f16 row-major. One block = 8 m-tiles x 1 n-tile; the 16xKp weight tile
// (contiguous, <=24KB) is staged into LDS once (async DMA when available) and
// shared by all 8 waves; A fragments stream from global with prefetch.
// ---------------------------------------------------------------------------
__global__ __launch_bounds__(256) void gemm_wmma_kernel(
    const _Float16* __restrict__ X, const _Float16* __restrict__ Wh,
    const float* __restrict__ scale, const float* __restrict__ bias,
    int M, int Kp, int Nc, int NcPad, int relu,
    _Float16* __restrict__ Yh, float* __restrict__ Yf, int ldOut) {
  __shared__ alignas(32) _Float16 sB[16 * 768];   // max Kp = 768 -> 24 KB

  int wave = threadIdx.x >> 5;
  int lane = threadIdx.x & 31;
  int mTiles = M >> 4;
  int mBlocks = (mTiles + 7) >> 3;
  int mBlk = blockIdx.x % mBlocks;
  int nT   = blockIdx.x / mBlocks;
  int n0 = nT << 4;

  // ---- stage 16 x Kp weight tile (contiguous rows n0..n0+15) into LDS ----
  {
    const char* src = (const char*)(Wh + (long long)n0 * Kp);
    char* dst = (char*)sB;
    int bytes = 32 * Kp;                       // 16 rows * Kp * 2B
    for (int o = threadIdx.x * 16; o < bytes; o += 256 * 16) {
#if HAVE_ASYNC_LDS
      __builtin_amdgcn_global_load_async_to_lds_b128(
          (__attribute__((address_space(1))) v4i*)(src + o),
          (__attribute__((address_space(3))) v4i*)(dst + o), 0, 0);
#else
      *(uint4*)(dst + o) = *(const uint4*)(src + o);
#endif
    }
#if HAVE_ASYNC_LDS
    __builtin_amdgcn_s_wait_asynccnt(0);
#endif
    __syncthreads();
  }

  int mT = (mBlk << 3) + wave;
  if (mT >= mTiles) return;                    // wave-uniform (after the barrier)
  int m0 = mT << 4;
  int r = lane & 15, half = lane >> 4;

  // A fragment: row (m0+r), 16 f16 at k0+half*16
  const _Float16* xrow = X + (long long)(m0 + r) * Kp + half * 16;
  // B fragment from LDS: column (n0+r) == LDS row r, 16 f16 at k0+half*16
  const _Float16* brow = sB + r * Kp + half * 16;

  v8f acc = {};
  int kTiles = Kp >> 5;
  for (int kt = 0; kt < kTiles; ++kt) {
    int k0 = kt << 5;
    if (kt + 1 < kTiles) __builtin_prefetch(xrow + k0 + 32, 0, 1);
    union { v16h v; v8h h[2]; } a;
    a.h[0] = *(const v8h*)(xrow + k0);
    a.h[1] = *(const v8h*)(xrow + k0 + 8);
    v16h bm = *(const v16h*)(brow + k0);
    acc = __builtin_amdgcn_wmma_f32_16x16x32_f16(
        /*neg_a=*/false, a.v, /*neg_b=*/false, bm,
        /*c_mod=*/(short)0, acc, /*reuse_a=*/false, /*reuse_b=*/false);
  }

  int col = n0 + r;
  float s = scale[col];
  float bb = bias[col];
  for (int i = 0; i < 8; ++i) {
    int row = m0 + i + half * 8;
    if (col < Nc) {
      float y = acc[i] * s + bb;
      if (relu) y = fmaxf(y, 0.f);
      if (Yf) Yf[(long long)row * ldOut + col] = y;
      else    Yh[(long long)row * ldOut + col] = (_Float16)y;
    }
  }
}

// ---------------------------------------------------------------------------
// Max-pool over nsample: in (groups*nsample, C) f16 -> out (groups, C) f16
// ---------------------------------------------------------------------------
__global__ void maxpool_kernel(const _Float16* __restrict__ in, int nsample, int C,
                               int groups, _Float16* __restrict__ out) {
  long long total = (long long)groups * C;
  long long gs = (long long)gridDim.x * blockDim.x;
  for (long long e = (long long)blockIdx.x * blockDim.x + threadIdx.x; e < total; e += gs) {
    long long g = e / C; int c = (int)(e % C);
    float m = -3.0e38f;
    const _Float16* base = in + (g * nsample) * (long long)C + c;
    for (int j = 0; j < nsample; ++j) m = fmaxf(m, (float)base[(long long)j * C]);
    out[e] = (_Float16)m;
  }
}

// ---------------------------------------------------------------------------
// 3-NN (smallest squared distances, first-occurrence tie-break) + norm weights
// ---------------------------------------------------------------------------
__global__ void knn3_kernel(const float* __restrict__ xyz1, int s1, int N1,
                            const float* __restrict__ xyz2, int N2,
                            int* __restrict__ idx3, float* __restrict__ w3, int total) {
  int gid = blockIdx.x * blockDim.x + threadIdx.x;
  if (gid >= total) return;
  int b = gid / N1;
  const float* p = xyz1 + (long long)gid * s1;
  const float* Q = xyz2 + (long long)b * N2 * 3;
  float d0 = 3e38f, d1 = 3e38f, d2 = 3e38f;
  int i0 = 0, i1 = 0, i2 = 0;
  for (int i = 0; i < N2; ++i) {
    float dx = Q[i * 3 + 0] - p[0];
    float dy = Q[i * 3 + 1] - p[1];
    float dz = Q[i * 3 + 2] - p[2];
    float d = dx * dx + dy * dy + dz * dz;
    if (d < d0)      { d2 = d1; i2 = i1; d1 = d0; i1 = i0; d0 = d; i0 = i; }
    else if (d < d1) { d2 = d1; i2 = i1; d1 = d; i1 = i; }
    else if (d < d2) { d2 = d; i2 = i; }
  }
  float w0 = 1.f / (d0 + 1e-8f), w1 = 1.f / (d1 + 1e-8f), w2 = 1.f / (d2 + 1e-8f);
  float ws = w0 + w1 + w2;
  idx3[gid * 3 + 0] = i0; idx3[gid * 3 + 1] = i1; idx3[gid * 3 + 2] = i2;
  w3[gid * 3 + 0] = w0 / ws; w3[gid * 3 + 1] = w1 / ws; w3[gid * 3 + 2] = w2 / ws;
}

// ---------------------------------------------------------------------------
// FP input build: [points1 (C1), interp(points2, idx3, w3) (C2), 0-pad to Kp]
// ---------------------------------------------------------------------------
__global__ void fp_build_kernel(const _Float16* __restrict__ p1, int C1,
                                const _Float16* __restrict__ p2, int C2,
                                const int* __restrict__ idx3, const float* __restrict__ w3,
                                int N1, int N2, long long rows, int Kp,
                                _Float16* __restrict__ out) {
  long long total = rows * Kp;
  long long gs = (long long)gridDim.x * blockDim.x;
  for (long long e = (long long)blockIdx.x * blockDim.x + threadIdx.x; e < total; e += gs) {
    int k = (int)(e % Kp);
    long long rr = e / Kp;
    float val = 0.f;
    if (k < C1) {
      val = (float)p1[rr * C1 + k];
    } else if (k < C1 + C2) {
      int c = k - C1;
      int b = (int)(rr / N1);
      const int*   id = idx3 + rr * 3;
      const float* w  = w3 + rr * 3;
      long long base = (long long)b * N2;
      val = w[0] * (float)p2[(base + id[0]) * C2 + c] +
            w[1] * (float)p2[(base + id[1]) * C2 + c] +
            w[2] * (float)p2[(base + id[2]) * C2 + c];
    }
    out[e] = (_Float16)val;
  }
}

// ---------------------------------------------------------------------------
// Host orchestration
// ---------------------------------------------------------------------------
static const int LC_OUT[23] = {32,32,64, 64,64,128, 128,128,256, 256,256,512,
                               256,256, 256,256, 256,128, 128,128,128, 128, 13};
static const int LC_IN[23]  = {9,32,32, 67,64,64, 131,128,128, 259,256,256,
                               768,256, 384,256, 320,256, 128,128,128, 128, 128};

extern "C" void kernel_launch(void* const* d_in, const int* in_sizes, int n_in,
                              void* d_out, int out_size, void* d_ws, size_t ws_size,
                              hipStream_t stream) {
  (void)in_sizes; (void)n_in; (void)out_size; (void)ws_size;
  const int B = 8, N0 = 4096;
  const float* xyz0 = (const float*)d_in[0];   // (8,4096,6)

  // ---- workspace carve (deterministic) ----
  char* ws = (char*)d_ws;
  size_t off = 0;
  auto alloc = [&](size_t bytes) -> char* {
    char* p = ws + off;
    off = (off + bytes + 255) & ~(size_t)255;
    return p;
  };
  _Float16* bufA = (_Float16*)alloc((size_t)262144 * 64 * 2);
  _Float16* bufB = (_Float16*)alloc((size_t)262144 * 64 * 2);
  _Float16* l1h  = (_Float16*)alloc((size_t)8192 * 64 * 2);
  _Float16* l2h  = (_Float16*)alloc((size_t)2048 * 128 * 2);
  _Float16* l3h  = (_Float16*)alloc((size_t)512 * 256 * 2);
  _Float16* l4h  = (_Float16*)alloc((size_t)128 * 512 * 2);
  _Float16* l3ph = (_Float16*)alloc((size_t)512 * 256 * 2);
  _Float16* l2ph = (_Float16*)alloc((size_t)2048 * 256 * 2);
  _Float16* l1ph = (_Float16*)alloc((size_t)8192 * 128 * 2);
  float* nxyz1 = (float*)alloc((size_t)8 * 1024 * 3 * 4);
  float* nxyz2 = (float*)alloc((size_t)8 * 256 * 3 * 4);
  float* nxyz3 = (float*)alloc((size_t)8 * 64 * 3 * 4);
  float* nxyz4 = (float*)alloc((size_t)8 * 16 * 3 * 4);
  int*   fpsIdx  = (int*)alloc((size_t)8 * 1024 * 4);
  int*   ballIdx = (int*)alloc((size_t)8 * 1024 * 32 * 4);
  float* distWs  = (float*)alloc((size_t)8 * 4096 * 4);
  int*   idx3    = (int*)alloc((size_t)8 * 4096 * 3 * 4);
  float* w3      = (float*)alloc((size_t)8 * 4096 * 3 * 4);

  struct LayerDev { int cout, cin, Kp, NcPad; _Float16* wh; float* scale; float* bias; };
  LayerDev L[23];
  for (int i = 0; i < 23; ++i) {
    L[i].cout = LC_OUT[i]; L[i].cin = LC_IN[i];
    L[i].Kp = (LC_IN[i] + 31) & ~31;
    L[i].NcPad = (LC_OUT[i] + 15) & ~15;
    L[i].wh = (_Float16*)alloc((size_t)L[i].NcPad * L[i].Kp * 2);
    L[i].scale = (float*)alloc((size_t)L[i].NcPad * 4);
    L[i].bias  = (float*)alloc((size_t)L[i].NcPad * 4);
  }

  auto gsBlocks = [](long long total) -> unsigned int {
    long long blk = (total + 255) / 256;
    if (blk < 1) blk = 1;
    if (blk > 1048576) blk = 1048576;
    return (unsigned int)blk;
  };

  // ---- weight prep ----
  for (int i = 0; i < 23; ++i) {
    const float *W, *b, *g = nullptr, *bt = nullptr, *m = nullptr, *v = nullptr;
    if (i < 22) {
      W  = (const float*)d_in[1 + 6 * i + 0];
      b  = (const float*)d_in[1 + 6 * i + 1];
      g  = (const float*)d_in[1 + 6 * i + 2];
      bt = (const float*)d_in[1 + 6 * i + 3];
      m  = (const float*)d_in[1 + 6 * i + 4];
      v  = (const float*)d_in[1 + 6 * i + 5];
    } else {
      W = (const float*)d_in[133];
      b = (const float*)d_in[134];
    }
    long long tot = (long long)L[i].NcPad * L[i].Kp;
    prep_layer_kernel<<<gsBlocks(tot), 256, 0, stream>>>(
        W, b, g, bt, m, v, L[i].cout, L[i].cin, L[i].Kp, L[i].NcPad,
        L[i].wh, L[i].scale, L[i].bias);
  }

  auto gemm = [&](const _Float16* X, int li, int M, _Float16* Yh, float* Yf, int relu) {
    int mTiles = M >> 4, nTiles = L[li].NcPad >> 4;
    int mBlocks = (mTiles + 7) >> 3;
    gemm_wmma_kernel<<<(unsigned int)(mBlocks * nTiles), 256, 0, stream>>>(
        X, L[li].wh, L[li].scale, L[li].bias,
        M, L[li].Kp, L[li].cout, L[li].NcPad, relu, Yh, Yf, L[li].cout);
  };

  // ---- set-abstraction stage ----
  auto sa_stage = [&](const float* xyz, int xs, int N, int npoint, float radius, int nsample,
                      const _Float16* pts16, const float* pts32, int Cp, int pstride,
                      int L0, float* newxyz, _Float16* pooled) {
    fps_kernel<<<B, 256, 0, stream>>>(xyz, xs, N, npoint, fpsIdx, newxyz, distWs);
    int centers = B * npoint;
    ball_query_kernel<<<(centers + 255) / 256, 256, 0, stream>>>(
        xyz, xs, N, newxyz, npoint, radius * radius, nsample, ballIdx, B);
    int Kp = L[L0].Kp;
    long long M = (long long)centers * nsample;
    group_kernel<<<gsBlocks(M * Kp), 256, 0, stream>>>(
        xyz, xs, pts16, pts32, Cp, pstride, newxyz, ballIdx,
        npoint, nsample, N, bufB, Kp, B);
    gemm(bufB, L0 + 0, (int)M, bufA, nullptr, 1);
    gemm(bufA, L0 + 1, (int)M, bufB, nullptr, 1);
    gemm(bufB, L0 + 2, (int)M, bufA, nullptr, 1);
    int C = L[L0 + 2].cout;
    maxpool_kernel<<<gsBlocks((long long)centers * C), 256, 0, stream>>>(
        bufA, nsample, C, centers, pooled);
  };

  // ---- feature-propagation stage ----
  auto fp_stage = [&](const float* xyz1, int s1, int N1, const float* xyz2, int N2,
                      const _Float16* pts1, int C1, const _Float16* pts2, int C2,
                      int L0, int nLayers, _Float16* outFinal) {
    int tot = B * N1;
    knn3_kernel<<<(tot + 255) / 256, 256, 0, stream>>>(xyz1, s1, N1, xyz2, N2, idx3, w3, tot);
    int Kp = L[L0].Kp;
    long long rows = tot;
    fp_build_kernel<<<gsBlocks(rows * Kp), 256, 0, stream>>>(
        pts1, C1, pts2, C2, idx3, w3, N1, N2, rows, Kp, bufB);
    const _Float16* cur = bufB;
    _Float16* nxt = bufA;
    for (int i = 0; i < nLayers; ++i) {
      _Float16* dst = (i == nLayers - 1) ? outFinal : nxt;
      gemm(cur, L0 + i, (int)rows, dst, nullptr, 1);
      cur = dst;
      nxt = (nxt == bufA) ? bufB : bufA;
    }
  };

  // ---- pipeline ----
  // SA1: xyz=l0_xyz (stride 6), points = full 6-ch input
  sa_stage(xyz0, 6, N0, 1024, 0.1f, 32, nullptr, xyz0, 6, 6, 0, nxyz1, l1h);
  // SA2
  sa_stage(nxyz1, 3, 1024, 256, 0.2f, 32, l1h, nullptr, 64, 64, 3, nxyz2, l2h);
  // SA3
  sa_stage(nxyz2, 3, 256, 64, 0.4f, 32, l2h, nullptr, 128, 128, 6, nxyz3, l3h);
  // SA4
  sa_stage(nxyz3, 3, 64, 16, 0.8f, 32, l3h, nullptr, 256, 256, 9, nxyz4, l4h);

  // FP4: l3 <- l3 ++ interp(l4 on l3_xyz)
  fp_stage(nxyz3, 3, 64, nxyz4, 16, l3h, 256, l4h, 512, 12, 2, l3ph);
  // FP3
  fp_stage(nxyz2, 3, 256, nxyz3, 64, l2h, 128, l3ph, 256, 14, 2, l2ph);
  // FP2
  fp_stage(nxyz1, 3, 1024, nxyz2, 256, l1h, 64, l2ph, 256, 16, 2, l1ph);
  // FP1: points1 = None; xyz1 = l0_xyz (stride 6). Final layer -> bufA.
  fp_stage(xyz0, 6, 4096, nxyz1, 1024, nullptr, 0, l1ph, 128, 18, 3, bufA);

  // head (BN+ReLU): bufA -> bufB
  gemm(bufA, 21, B * N0, bufB, nullptr, 1);
  // conv2 (plain linear), f32 output to d_out (8*4096*13)
  gemm(bufB, 22, B * N0, nullptr, (float*)d_out, 0);
}
